// GINLayer_56530359550254
// MI455X (gfx1250) — compile-verified, matrix operand order checked by reading, not run on hardware
//
#include <hip/hip_runtime.h>
#include <hip/hip_bf16.h>

#define GIN_N  100000
#define GIN_E  1600000
#define GIN_D  32
#define GIN_C  2
#define GIN_NL 4
#define GIN_G  512
// 1/sqrt(1 + 1e-3)
#define BN_RS  0.99950037f

#define MLP_BLOCKS 256   // persistent blocks; 8 waves each -> 2048 waves over 6250 tiles

typedef __attribute__((ext_vector_type(16))) _Float16 v16h;
typedef __attribute__((ext_vector_type(8)))  float    v8f;

// ---------------------------------------------------------------------------
// Edge aggregation: one wave32 per edge, lane = feature dim.
// Coalesced 128B row gather (L2-resident) + 32 non-returning f32 atomics.
// ---------------------------------------------------------------------------
__global__ __launch_bounds__(256) void gin_edge_agg(
    const float* __restrict__ h, const int* __restrict__ src,
    const int* __restrict__ dst, float* __restrict__ agg)
{
    const int lane = threadIdx.x & 31;
    const int e    = (int)((blockIdx.x * 256u + threadIdx.x) >> 5);
    if (e >= GIN_E) return;
    const int s = src[e];
    const int d = dst[e];
    const float v = h[(size_t)s * GIN_D + lane];
    atomicAdd(&agg[(size_t)d * GIN_D + lane], v);
}

// ---------------------------------------------------------------------------
// Fused GIN node update (persistent waves, grid-stride over 16-node tiles):
//   z = h + agg; y = relu(bn1(z@W1+b1)); h' = relu(norm(relu(bn2(y@W2+b2))))
// WMMA f32_16x16x32_f16 for both GEMMs; weight B-fragments + folded BN
// constants hoisted out of the tile loop. Pooling fused into the epilogue.
// ---------------------------------------------------------------------------
__global__ __launch_bounds__(256) void gin_mlp_wmma(
    const float* __restrict__ h_in,  const float* __restrict__ agg,
    const float* __restrict__ W1,    const float* __restrict__ b1,
    const float* __restrict__ bn1g,  const float* __restrict__ bn1b,
    const float* __restrict__ W2,    const float* __restrict__ b2,
    const float* __restrict__ bn2g,  const float* __restrict__ bn2b,
    const float* __restrict__ ng,    const float* __restrict__ nb,
    const int*   __restrict__ batch,
    float* __restrict__ h_out, float* __restrict__ pool_lvl)
{
    __shared__ _Float16 ysh[8][16][GIN_D];   // per-wave 16x32 f16 transpose tile

    const int lane   = threadIdx.x & 31;
    const int wid    = threadIdx.x >> 5;
    const int nwaves = gridDim.x * 8;
    const int n      = lane & 15;            // column (B/C fragments)
    const int hi     = lane >> 4;
    const int kb     = hi * 8;               // K-base for this lane group
    const int m      = n;                    // row (A fragment)
    const int q0     = kb >> 2;              // float4 index of K-base (0 or 2)

    // ---- hoisted: weight B-fragments (ISA 16-bit B layout) + BN constants ----
    v16h w1f[2], w2f[2];
    float s1[2], o1[2], s2[2], o2[2], sn[2], on[2];
    #pragma unroll
    for (int ct = 0; ct < 2; ++ct) {
        const int col = ct * 16 + n;
        #pragma unroll
        for (int j = 0; j < 8; ++j) {
            w1f[ct][j]     = (_Float16)W1[(kb + j)      * GIN_D + col];
            w1f[ct][8 + j] = (_Float16)W1[(kb + 16 + j) * GIN_D + col];
            w2f[ct][j]     = (_Float16)W2[(kb + j)      * GIN_D + col];
            w2f[ct][8 + j] = (_Float16)W2[(kb + 16 + j) * GIN_D + col];
        }
        s1[ct] = bn1g[col] * BN_RS;  o1[ct] = b1[col] * s1[ct] + bn1b[col];
        s2[ct] = bn2g[col] * BN_RS;  o2[ct] = b2[col] * s2[ct] + bn2b[col];
        sn[ct] = ng[col]   * BN_RS;  on[ct] = nb[col];
    }

    for (int tile = blockIdx.x * 8 + wid; tile < GIN_N / 16; tile += nwaves) {
        const int row0 = tile * 16;

        // ---- A fragment: z = h + agg (row m), vectorized float4 loads ----
        const float4* hz = (const float4*)(h_in + (size_t)(row0 + m) * GIN_D);
        const float4* az = (const float4*)(agg  + (size_t)(row0 + m) * GIN_D);
        const float4 ha0 = hz[q0],     ha1 = hz[q0 + 1];   // K = kb..kb+7
        const float4 hb0 = hz[q0 + 4], hb1 = hz[q0 + 5];   // K = kb+16..kb+23
        const float4 aa0 = az[q0],     aa1 = az[q0 + 1];
        const float4 ab0 = az[q0 + 4], ab1 = az[q0 + 5];
        v16h a;
        a[0]  = (_Float16)(ha0.x + aa0.x);  a[1]  = (_Float16)(ha0.y + aa0.y);
        a[2]  = (_Float16)(ha0.z + aa0.z);  a[3]  = (_Float16)(ha0.w + aa0.w);
        a[4]  = (_Float16)(ha1.x + aa1.x);  a[5]  = (_Float16)(ha1.y + aa1.y);
        a[6]  = (_Float16)(ha1.z + aa1.z);  a[7]  = (_Float16)(ha1.w + aa1.w);
        a[8]  = (_Float16)(hb0.x + ab0.x);  a[9]  = (_Float16)(hb0.y + ab0.y);
        a[10] = (_Float16)(hb0.z + ab0.z);  a[11] = (_Float16)(hb0.w + ab0.w);
        a[12] = (_Float16)(hb1.x + ab1.x);  a[13] = (_Float16)(hb1.y + ab1.y);
        a[14] = (_Float16)(hb1.z + ab1.z);  a[15] = (_Float16)(hb1.w + ab1.w);

        // ---- GEMM1 + BN1 + ReLU -> f16 LDS tile (layout transpose) ----
        #pragma unroll
        for (int ct = 0; ct < 2; ++ct) {
            const int col = ct * 16 + n;
            v8f c = {};
            c = __builtin_amdgcn_wmma_f32_16x16x32_f16(false, a, false, w1f[ct],
                                                       (short)0, c, false, false);
            #pragma unroll
            for (int r = 0; r < 8; ++r) {
                float v = c[r] * s1[ct] + o1[ct];
                v = v > 0.0f ? v : 0.0f;
                ysh[wid][r + 8 * hi][col] = (_Float16)v;   // row r+8*hi
            }
        }

        // ---- Reload as A fragment (intra-wave LDS, DS ops in-order) ----
        v16h a2;
        #pragma unroll
        for (int j = 0; j < 8; ++j) {
            a2[j]     = ysh[wid][m][kb + j];
            a2[8 + j] = ysh[wid][m][kb + 16 + j];
        }

        // graph ids for this tile's rows (shared by both column halves)
        int bg[8];
        #pragma unroll
        for (int r = 0; r < 8; ++r) bg[r] = batch[row0 + r + 8 * hi];

        // ---- GEMM2 + BN2/ReLU + outer norm/ReLU + store + fused pooling ----
        #pragma unroll
        for (int ct = 0; ct < 2; ++ct) {
            const int col = ct * 16 + n;
            v8f c = {};
            c = __builtin_amdgcn_wmma_f32_16x16x32_f16(false, a2, false, w2f[ct],
                                                       (short)0, c, false, false);
            #pragma unroll
            for (int r = 0; r < 8; ++r) {
                float v = c[r] * s2[ct] + o2[ct];
                v = v > 0.0f ? v : 0.0f;
                v = v * sn[ct] + on[ct];
                v = v > 0.0f ? v : 0.0f;
                const int row = row0 + r + 8 * hi;
                h_out[(size_t)row * GIN_D + col] = v;
                atomicAdd(&pool_lvl[(size_t)bg[r] * GIN_D + col], v);
            }
        }
    }
}

// ---------------------------------------------------------------------------
// Level-0 pooling: pool0[batch[node]] += node_embeddings[node]
// ---------------------------------------------------------------------------
__global__ __launch_bounds__(256) void gin_pool0(
    const float* __restrict__ emb, const int* __restrict__ batch,
    float* __restrict__ pool0)
{
    const int i = blockIdx.x * 256 + threadIdx.x;
    if (i >= GIN_N * GIN_D) return;
    const int node = i >> 5;
    const int d    = i & 31;
    atomicAdd(&pool0[(size_t)batch[node] * GIN_D + d], emb[i]);
}

// ---------------------------------------------------------------------------
// Readout: score[g] = sum_l pool[l][g] @ dense_W[l] + dense_b[l]; softmax(C=2)
// ---------------------------------------------------------------------------
__global__ __launch_bounds__(256) void gin_readout(
    const float* __restrict__ pool, const float* __restrict__ dW,
    const float* __restrict__ db, float* __restrict__ out)
{
    const int g = blockIdx.x * 256 + threadIdx.x;
    if (g >= GIN_G) return;
    float s0 = 0.0f, s1 = 0.0f;
    for (int l = 0; l < GIN_NL + 1; ++l) {
        const float* p = pool + (size_t)l * GIN_G * GIN_D + (size_t)g * GIN_D;
        const float* w = dW + l * GIN_D * GIN_C;
        #pragma unroll
        for (int d = 0; d < GIN_D; ++d) {
            const float pv = p[d];
            s0 = fmaf(pv, w[d * 2 + 0], s0);
            s1 = fmaf(pv, w[d * 2 + 1], s1);
        }
        s0 += db[l * 2 + 0];
        s1 += db[l * 2 + 1];
    }
    const float mx  = fmaxf(s0, s1);
    const float e0  = __expf(s0 - mx);
    const float e1  = __expf(s1 - mx);
    const float inv = 1.0f / (e0 + e1);
    out[g * 2 + 0] = e0 * inv;
    out[g * 2 + 1] = e1 * inv;
}

extern "C" void kernel_launch(void* const* d_in, const int* in_sizes, int n_in,
                              void* d_out, int out_size, void* d_ws, size_t ws_size,
                              hipStream_t stream) {
    (void)in_sizes; (void)n_in; (void)out_size; (void)ws_size;

    const float* emb   = (const float*)d_in[0];
    const int*   eidx  = (const int*)  d_in[1];   // [2, E]: src then dst
    const int*   batch = (const int*)  d_in[2];
    const float* W1    = (const float*)d_in[3];
    const float* b1    = (const float*)d_in[4];
    const float* bn1g  = (const float*)d_in[5];
    const float* bn1b  = (const float*)d_in[6];
    const float* W2    = (const float*)d_in[7];
    const float* b2    = (const float*)d_in[8];
    const float* bn2g  = (const float*)d_in[9];
    const float* bn2b  = (const float*)d_in[10];
    const float* ng    = (const float*)d_in[11];
    const float* nb    = (const float*)d_in[12];
    const float* dW    = (const float*)d_in[13];
    const float* db    = (const float*)d_in[14];

    const size_t ND = (size_t)GIN_N * GIN_D;
    float* hA   = (float*)d_ws;
    float* hB   = hA + ND;
    float* agg  = hB + ND;
    float* pool = agg + ND;                    // (NL+1) * G * D floats

    // zero pooling accumulators once per launch
    hipMemsetAsync(pool, 0,
                   (size_t)(GIN_NL + 1) * GIN_G * GIN_D * sizeof(float), stream);

    // hidden level 0 pooling (input embeddings)
    gin_pool0<<<(GIN_N * GIN_D + 255) / 256, 256, 0, stream>>>(emb, batch, pool);

    const float* hcur = emb;
    float* bufs[2] = { hA, hB };

    const int edge_blocks = (GIN_E * 32 + 255) / 256;       // 1 wave per edge

    for (int i = 0; i < GIN_NL; ++i) {
        hipMemsetAsync(agg, 0, ND * sizeof(float), stream);
        gin_edge_agg<<<edge_blocks, 256, 0, stream>>>(hcur, eidx, eidx + GIN_E, agg);

        float* hnext = bufs[i & 1];
        gin_mlp_wmma<<<MLP_BLOCKS, 256, 0, stream>>>(
            hcur, agg,
            W1 + i * GIN_D * GIN_D, b1 + i * GIN_D,
            bn1g + i * GIN_D, bn1b + i * GIN_D,
            W2 + i * GIN_D * GIN_D, b2 + i * GIN_D,
            bn2g + i * GIN_D, bn2b + i * GIN_D,
            ng + i * GIN_D, nb + i * GIN_D,
            batch, hnext,
            pool + (size_t)(i + 1) * GIN_G * GIN_D);
        hcur = hnext;
    }

    gin_readout<<<(GIN_G + 255) / 256, 256, 0, stream>>>(pool, dW, db, (float*)d_out);
}
